// SimpleLSTMWithAttention_20048907338272
// MI455X (gfx1250) — matrix operand order repaired and checked by reading, compiled, not run on hardware
//
#include <hip/hip_runtime.h>
#include <hip/hip_bf16.h>
#include <stdint.h>
#include <stddef.h>

// Problem constants
#define NB   32        // batch
#define LSEQ 512       // sequence length
#define EMB  512       // embedding dim
#define HID  1024      // hidden dim
#define KTOT 1536      // H + E (concat K)
#define GTOT 4096      // 4H gate columns
#define NWG  64        // workgroups in recurrent kernel (1024/16 units)
#define TPB  256       // threads per block (8 wave32)
#define NTILE (KTOT / 32)                  // 48 K-tiles per step

// LDS layout (dynamic shared)
#define SW_BYTES  (64 * KTOT * 2)          // 196608: 64 weight cols (4 gates x 16 units), bf16, K-contig
#define SG_OFF    SW_BYTES                 // 4*32*16 fp32 gate pre-activations = 8192 B
#define SC_OFF    (SW_BYTES + 8192)        // 32*16 fp32 cell state = 2048 B
#define SMEM_BYTES (SW_BYTES + 8192 + 2048)

// Workspace layout (bytes)
#define WS_WT 0                            // bf16 W^T (4096 x 1536)   = 12,582,912 B
#define WS_XE 12582912                     // bf16 Xe (L, N, E)        = 16,777,216 B
#define WS_H0 29360128                     // bf16 h ping (32x1024)    = 65,536 B
#define WS_H1 29425664                     // bf16 h pong              = 65,536 B
#define WS_SY 29491200                     // 2 x u32 barrier state

typedef __attribute__((ext_vector_type(16))) __bf16   v16bf;
typedef __attribute__((ext_vector_type(8)))  float    v8f;
typedef __attribute__((ext_vector_type(4)))  unsigned v4u;
typedef __attribute__((ext_vector_type(8)))  int      v8i;
typedef __attribute__((ext_vector_type(4)))  int      v4i;

#if __has_builtin(__builtin_amdgcn_tensor_load_to_lds)
#define HAVE_TDM 1
#endif

union Frag { v16bf v; unsigned u[8]; };

__device__ __forceinline__ unsigned short f2bf(float f) {
  unsigned u = __builtin_bit_cast(unsigned, f);
  u += 0x7FFFu + ((u >> 16) & 1u);         // round-to-nearest-even
  return (unsigned short)(u >> 16);
}

__device__ __forceinline__ float sigmoidf_(float x) {
  return 1.0f / (1.0f + __expf(-x));
}

// One K-tile of fragments: A = two b128 global loads, B = LDS dwords {0..3} and {8..11}.
__device__ __forceinline__ void load_tile(Frag& a, Frag& b,
                                          const unsigned* __restrict__ pa,
                                          const unsigned* __restrict__ pb) {
#pragma unroll
  for (int v = 0; v < 8; ++v) {
    a.u[v] = (v < 4) ? pa[v] : pa[v + 4];
    b.u[v] = (v < 4) ? pb[v] : pb[v + 4];
  }
}

// Device-wide split barrier (all NWG workgroups co-resident: 64 WGs x ~200KB
// LDS = 1 WG/WGP, trivially resident on MI455X).
__device__ __forceinline__ void grid_sync(unsigned* cnt, unsigned* gen) {
  __threadfence();
  __syncthreads();
  if (threadIdx.x == 0) {
    unsigned g = __atomic_load_n(gen, __ATOMIC_RELAXED);
    unsigned a = atomicAdd(cnt, 1u);
    if (a == NWG - 1) {
      __atomic_store_n(cnt, 0u, __ATOMIC_RELAXED);
      __atomic_fetch_add(gen, 1u, __ATOMIC_RELEASE);
    } else {
      while (__atomic_load_n(gen, __ATOMIC_ACQUIRE) == g)
        __builtin_amdgcn_s_sleep(2);
    }
  }
  __syncthreads();
}

// ---------------- prep kernels ----------------

__global__ void init_kernel(unsigned short* h0, unsigned short* h1, unsigned* sync) {
  for (int i = threadIdx.x; i < NB * HID; i += TPB) { h0[i] = 0; h1[i] = 0; }
  if (threadIdx.x < 2) sync[threadIdx.x] = 0;
}

// W^T bf16: Wt[c][r] = bf16(Ww[r][c]); K-contiguous per output column
__global__ void wt_kernel(const float* __restrict__ Ww, unsigned short* __restrict__ Wt) {
  int c = blockIdx.x;
  for (int r = threadIdx.x; r < KTOT; r += TPB)
    Wt[(size_t)c * KTOT + r] = f2bf(Ww[(size_t)r * GTOT + c]);
}

// Xe[t][n][e] = bf16(Emb[X[n][t]][e])   (time-major so each step is contiguous)
__global__ void embed_kernel(const int* __restrict__ X, const float* __restrict__ Emb,
                             unsigned short* __restrict__ Xe) {
  int bid = blockIdx.x;
  int t = bid & (LSEQ - 1);
  int n = bid >> 9;
  int tok = X[n * LSEQ + t];
  const float* row = Emb + (size_t)tok * EMB;
  unsigned short* dst = Xe + ((size_t)t * NB + n) * EMB;
  for (int e = threadIdx.x; e < EMB; e += TPB) dst[e] = f2bf(row[e]);
}

// ---------------- persistent recurrent kernel ----------------

__global__ void lstm_kernel(const unsigned short* __restrict__ Wt,
                            const unsigned short* __restrict__ Xe,
                            const float* __restrict__ Wb,
                            unsigned short* __restrict__ h0,
                            unsigned short* __restrict__ h1,
                            unsigned* __restrict__ sync,
                            float* __restrict__ out) {
  extern __shared__ char smem[];
  unsigned short* sW = (unsigned short*)smem;            // [64 cols][1536 K] bf16
  float*          sG = (float*)(smem + SG_OFF);          // [4 gates][32 batch][16 units]
  float*          sC = (float*)(smem + SC_OFF);          // [32 batch][16 units]

  const int tid  = threadIdx.x;
  const int u0   = blockIdx.x * 16;                      // owned hidden units [u0, u0+16)

  // ---- Stage this WG's weight slice into LDS once.
#ifdef HAVE_TDM
  // Tensor Data Mover: one 2D-tile descriptor per gate strip
  // (16 rows x 1536 bf16, row stride 1536 elem), issued by wave 0 only
  // (TDM ignores EXEC; keep it out of other waves' instruction streams).
  if (tid < 32) {
    unsigned ldsbase = (unsigned)(uintptr_t)(void*)sW;   // flat LDS addr[31:0] = LDS offset
#pragma unroll
    for (int q = 0; q < 4; ++q) {
      unsigned long long gaddr = (unsigned long long)(uintptr_t)Wt
                               + (unsigned long long)(q * HID + u0) * (KTOT * 2);
      v4u g0 = { 1u,                                            // count=1, user D#
                 ldsbase + (unsigned)q * (16u * KTOT * 2u),     // lds_addr
                 (unsigned)(gaddr & 0xFFFFFFFFu),               // global_addr[31:0]
                 (unsigned)((gaddr >> 32) & 0x1FFFFFFu) | (2u << 30) }; // addr[56:32] | type=2
      v8i g1 = { (int)(1u << 16),          // workgroup_mask=0, data_size=1 (2 bytes)
                 (int)(KTOT << 16),        // tensor_dim0 = 1536 (bits 79:48, low half)
                 (int)(16 << 16),          // tensor_dim0 hi=0, tensor_dim1 = 16
                 (int)(KTOT << 16),        // tensor_dim1 hi=0, tile_dim0 = 1536
                 16,                       // tile_dim1 = 16, tile_dim2 = 0
                 KTOT,                     // tensor_dim0_stride[31:0] = 1536
                 0, 0 };                   // stride hi / tensor_dim1_stride = 0 (2D)
      v4i z4 = {0, 0, 0, 0};
#if defined(__clang_major__) && (__clang_major__ >= 23)
      v8i z8 = {0, 0, 0, 0, 0, 0, 0, 0};
      __builtin_amdgcn_tensor_load_to_lds(g0, g1, z4, z4, z8, 0);
#else
      __builtin_amdgcn_tensor_load_to_lds(g0, g1, z4, z4, 0);
#endif
    }
    __builtin_amdgcn_s_wait_tensorcnt(0);
  }
#else
  // Fallback: cooperative dword copy (2 bf16 per dword, K-contiguous).
  for (int i = tid; i < 64 * (KTOT / 2); i += TPB) {
    int cl = i / (KTOT / 2), kd = i % (KTOT / 2);
    int q = cl >> 4, n = cl & 15;
    int col = q * HID + u0 + n;
    ((unsigned*)sW)[i] = *(const unsigned*)(Wt + (size_t)col * KTOT + kd * 2);
  }
#endif
  for (int i = tid; i < NB * 16; i += TPB) sC[i] = 0.0f;
  __syncthreads();

  // Wave -> tile assignment
  const int wv    = tid >> 5;          // 0..7
  const int lane  = tid & 31;
  const int mt    = wv >> 2;           // batch half (rows 16*mt ..)
  const int q     = wv & 3;            // gate: 0=f 1=i 2=o 3=g
  const int n     = lane & 15;         // tile column (unit)
  const int khalf = lane >> 4;         // K-half / M-half selector
  const int batchA = mt * 16 + n;      // A-fragment row (batch) for this lane
  const float bias = Wb[q * HID + u0 + n];

  // Per-lane invariant bases (lane K-skew folded in once):
  //   A tile loads = base + k0 elems -> dwords {0..3} and {8..11} (b128 pair)
  //   B tile loads = bW + 16*tile -> LDS dwords {0..3} and {8..11}
  const unsigned* bW = (const unsigned*)sW + (q * 16 + n) * (KTOT / 2) + 4 * khalf;
  const size_t    aRowH = (size_t)batchA * HID + 8 * khalf;
  const size_t    aRowX = (size_t)batchA * EMB + 8 * khalf;

  for (int t = 0; t < LSEQ; ++t) {
    const unsigned short* hprev = (t & 1) ? h1 : h0;
    unsigned short*       hnext = (t & 1) ? h0 : h1;
    const unsigned short* xet   = Xe + (size_t)t * NB * EMB;

    const unsigned short* aH = hprev + aRowH;
    const unsigned short* aX = xet + aRowX;

    v8f acc = {0.f, 0.f, 0.f, 0.f, 0.f, 0.f, 0.f, 0.f};

    // Depth-2 software pipeline over the 48 K-tiles (32 h-tiles then 16
    // x-tiles): loads for tile i+2 issue into a third buffer while the WMMA
    // consumes tile i, so two tiles of global+LDS latency stay hidden and the
    // per-tile wait only covers loads issued two tiles earlier. Fully
    // unrolled, so the h/x source select folds away at compile time.
    Frag A[3], B[3];
    load_tile(A[0], B[0], (const unsigned*)aH, bW);
    load_tile(A[1], B[1], (const unsigned*)(aH + 32), bW + 16);
#pragma unroll
    for (int ti = 0; ti < NTILE; ++ti) {
      if (ti + 2 < NTILE) {
        int tn = ti + 2;
        const unsigned* pa = (tn < 32) ? (const unsigned*)(aH + tn * 32)
                                       : (const unsigned*)(aX + (tn - 32) * 32);
        load_tile(A[(ti + 2) % 3], B[(ti + 2) % 3], pa, bW + tn * 16);
      }
      acc = __builtin_amdgcn_wmma_f32_16x16x32_bf16(
          false, A[ti % 3].v, false, B[ti % 3].v, (short)0, acc, false, false);
    }

    // C/D layout: lane = column n, VGPR r -> row r + 8*(lane>>4)
#pragma unroll
    for (int r = 0; r < 8; ++r) {
      int bt = mt * 16 + r + 8 * khalf;
      sG[q * 512 + bt * 16 + n] = acc[r] + bias;
    }
    __syncthreads();

    // Elementwise LSTM cell update for this WG's 32x16 (batch, unit) block.
    for (int e = tid; e < 512; e += TPB) {
      int bt = e >> 4, uu = e & 15;
      float fg = sigmoidf_(sG[e]);
      float ig = sigmoidf_(sG[512 + e]);
      float og = sigmoidf_(sG[1024 + e]);
      float gg = tanhf(sG[1536 + e]);
      float c  = fg * sC[e] + ig * gg;
      float h  = og * tanhf(c);
      sC[e] = c;
      int u = u0 + uu;
      out[(size_t)bt * LSEQ * HID + (size_t)t * HID + u] = h;        // H_all
      hnext[bt * HID + u] = f2bf(h);
      if (t == LSEQ - 1)
        out[(size_t)NB * LSEQ * HID + (size_t)bt * HID + u] = h;     // h_last
    }

    // Pull next step's x block toward the WGP while we sit in the barrier.
    if (t + 1 < LSEQ) {
      const char* nx = (const char*)(Xe + (size_t)(t + 1) * NB * EMB);
      __builtin_prefetch(nx + tid * 128, 0, 3);
    }

    grid_sync(sync, sync + 1);  // release h_next to all WGs
  }
}

// ---------------- host launch ----------------

extern "C" void kernel_launch(void* const* d_in, const int* in_sizes, int n_in,
                              void* d_out, int out_size, void* d_ws, size_t ws_size,
                              hipStream_t stream) {
  (void)in_sizes; (void)n_in; (void)out_size; (void)ws_size;
  const int*   X   = (const int*)d_in[0];
  const float* Emb = (const float*)d_in[1];
  const float* Ww  = (const float*)d_in[2];
  const float* Wb  = (const float*)d_in[3];
  float* out = (float*)d_out;

  char* ws = (char*)d_ws;
  unsigned short* Wt = (unsigned short*)(ws + WS_WT);
  unsigned short* Xe = (unsigned short*)(ws + WS_XE);
  unsigned short* h0 = (unsigned short*)(ws + WS_H0);
  unsigned short* h1 = (unsigned short*)(ws + WS_H1);
  unsigned* sync     = (unsigned*)(ws + WS_SY);

  (void)hipFuncSetAttribute((const void*)lstm_kernel,
                            hipFuncAttributeMaxDynamicSharedMemorySize, SMEM_BYTES);

  hipLaunchKernelGGL(init_kernel,  dim3(1),         dim3(TPB), 0, stream, h0, h1, sync);
  hipLaunchKernelGGL(wt_kernel,    dim3(GTOT),      dim3(TPB), 0, stream, Ww, Wt);
  hipLaunchKernelGGL(embed_kernel, dim3(NB * LSEQ), dim3(TPB), 0, stream, X, Emb, Xe);
  hipLaunchKernelGGL(lstm_kernel,  dim3(NWG), dim3(TPB), SMEM_BYTES, stream,
                     Wt, Xe, Wb, h0, h1, sync, out);
}
